// BipartiteTransformer_21861383537288
// MI455X (gfx1250) — matrix operand order repaired and checked by reading, not compiled, not run on hardware
//
#include <hip/hip_runtime.h>
#include <hip/hip_bf16.h>

// ---------------- problem constants (from reference) ----------------
#define NN      50000
#define EE      500000
#define IN_C    128
#define EDGE_C  64
#define HID     256
#define OUT_C   128
#define SEQ     100
#define HEADS   2
#define HEAD_D  128
#define EPS     1e-5f

typedef __attribute__((ext_vector_type(16))) __bf16 v16bf;
typedef __attribute__((ext_vector_type(8)))  __bf16 v8bf;
typedef __attribute__((ext_vector_type(8)))  float  v8f;

__device__ __forceinline__ unsigned short f2bf(float f) {
  unsigned u = __float_as_uint(f);
  u += 0x7FFFu + ((u >> 16) & 1u);
  return (unsigned short)(u >> 16);
}
__device__ __forceinline__ float leaky(float x) { return x > 0.f ? x : 0.2f * x; }
__device__ __forceinline__ int fmap_ord(float x) {            // monotone float->int map
  int i = __float_as_int(x);
  return i >= 0 ? i : (i ^ 0x7fffffff);
}
__device__ __forceinline__ float funmap_ord(int i) {
  return __int_as_float(i >= 0 ? i : (i ^ 0x7fffffff));
}
#define MAPNEGINF 0x807fffff   // fmap_ord(-inf)

// A-tile LDS row stride in ushorts: 128 elems + 8 pad -> 272B rows
// (keeps 16B alignment for b128 ops, stride = 68 dwords so half-wave
//  rows land on distinct banks)
#define ASTRIDE 136

// ================= K0: fold weights, pack B for WMMA =================
// v = w_gat_edge @ att_edge ; u = w_edge @ v ; c0 = b_edge.v
// bc = b_node @ w_gat ; Wc = w_node @ w_gat  (packed bf16, WMMA B layout)
__global__ void k_precompute(const float* __restrict__ w_node, const float* __restrict__ b_node,
                             const float* __restrict__ w_edge, const float* __restrict__ b_edge,
                             const float* __restrict__ w_gat,  const float* __restrict__ w_gat_edge,
                             const float* __restrict__ att_edge,
                             float* vbuf, float* ubuf, float* c0buf, float* bcbuf,
                             unsigned short* bpack) {
  int t = threadIdx.x;                                   // 256 threads, 1 block
  float a = 0.f;
  for (int n = 0; n < HID; ++n) a += w_gat_edge[t * HID + n] * att_edge[n];
  vbuf[t] = a;
  __syncthreads();
  if (t < EDGE_C) {
    float s = 0.f;
    for (int j = 0; j < HID; ++j) s += w_edge[t * HID + j] * vbuf[j];
    ubuf[t] = s;
  }
  if (t == 0) {
    float s = 0.f;
    for (int j = 0; j < HID; ++j) s += b_edge[j] * vbuf[j];
    c0buf[0] = s;
  }
  { float s = 0.f;
    for (int j = 0; j < HID; ++j) s += b_node[j] * w_gat[j * HID + t];
    bcbuf[t] = s; }
  // Wc[k][n], k<128, n<256 -> B layout: lane = (k%32>=16)<<4 | (n&15), slot = k&15
  for (int idx = t; idx < IN_C * HID; idx += 256) {
    int k = idx >> 8, n = idx & 255;
    float s = 0.f;
    for (int j = 0; j < HID; ++j) s += w_node[k * HID + j] * w_gat[j * HID + n];
    int ks   = k >> 5;            // k-step (32 wide)
    int w    = k & 31;
    int lane = ((w >> 4) << 4) | (n & 15);
    int slot = w & 15;
    int ccg  = n >> 4;            // 16-col chunk
    bpack[(((size_t)ks * 16 + ccg) * 32 + lane) * 16 + slot] = f2bf(s);
  }
}

// ================= K1: x -> bf16 =================
__global__ void k_cvt_x(const float* __restrict__ x, unsigned short* __restrict__ xb, int n) {
  int i = blockIdx.x * blockDim.x + threadIdx.x;
  if (i < n) xb[i] = f2bf(x[i]);
}

// ================= K2: init accumulators =================
__global__ void k_init(float* s_attr, float* deg, int* mbuf,
                       float* bn_sum, float* bn_sumsq, float* pooled) {
  int i = blockIdx.x * blockDim.x + threadIdx.x;
  if (i < NN * EDGE_C) s_attr[i] = 0.f;
  if (i < NN) { deg[i] = 0.f; mbuf[i] = MAPNEGINF; }
  if (i < HID) { bn_sum[i] = 0.f; bn_sumsq[i] = 0.f; pooled[i] = 0.f; }
}

// ================= K3: WMMA GEMM  xl = x_bf16 @ Wc + bc =================
// block = 128 thr (4 waves); block tile = 64 rows x 256 cols.
// A (64x128 bf16) staged once per block into LDS via ASYNC loads,
// consumed by all 4 waves; wave w computes 64 x cols[64w,64w+64).
// __launch_bounds__(128, 1): allow ~190 VGPRs so the 16 f32 accumulator
// tiles stay in registers (no scratch spills); still ~5 waves/SIMD.
__global__ void __launch_bounds__(128, 1)
k_gemm_xl(const unsigned short* __restrict__ xb,
          const unsigned short* __restrict__ bpack,
          const float* __restrict__ bc,
          float* __restrict__ xl) {
  __shared__ unsigned short sA[64 * ASTRIDE];            // 17.4 KB
  int wave = threadIdx.x >> 5;
  int lane = threadIdx.x & 31;
  int lrow = lane & 15, lhi = lane >> 4;
  int mrow0 = blockIdx.x * 64;

  // ---- async stage A tile: 64 rows x 256B = 1024 x 16B chunks ----
  const unsigned short* gbase = xb + (size_t)mrow0 * IN_C;
#pragma unroll
  for (int i = 0; i < 8; ++i) {
    int chunk = threadIdx.x + 128 * i;                   // 0..1023
    int row = chunk >> 4, seg = chunk & 15;
    const unsigned short* g = gbase + row * IN_C + seg * 8;
    unsigned lds_off = (unsigned)(size_t)(const void*)&sA[row * ASTRIDE + seg * 8];
    asm volatile("global_load_async_to_lds_b128 %0, %1, off"
                 :: "v"(lds_off), "v"(g) : "memory");
  }
  asm volatile("s_wait_asynccnt 0" ::: "memory");
  __syncthreads();

  // ---- compute ----
  v8f acc[4][4] = {};                                    // [m-sub][n-sub]
#pragma unroll
  for (int s = 0; s < 4; ++s) {                          // K steps of 32
    int ka = s * 32 + lhi * 16;
    v16bf a[4];
#pragma unroll
    for (int ms = 0; ms < 4; ++ms) {
      const unsigned short* ar = &sA[(ms * 16 + lrow) * ASTRIDE];
      v8bf alo = *reinterpret_cast<const v8bf*>(ar + ka);
      v8bf ahi = *reinterpret_cast<const v8bf*>(ar + ka + 8);
#pragma unroll
      for (int i = 0; i < 8; ++i) { a[ms][i] = alo[i]; a[ms][8 + i] = ahi[i]; }
    }
#pragma unroll
    for (int cc = 0; cc < 4; ++cc) {
      int ccg = wave * 4 + cc;
      v16bf b = *reinterpret_cast<const v16bf*>(
          bpack + (((size_t)s * 16 + ccg) * 32 + lane) * 16);
#pragma unroll
      for (int ms = 0; ms < 4; ++ms)
        acc[ms][cc] = __builtin_amdgcn_wmma_f32_16x16x32_bf16(
            false, a[ms], false, b, (short)0, acc[ms][cc], false, false);
    }
  }

  // ---- epilogue: +bc, store (guard tail rows) ----
#pragma unroll
  for (int ms = 0; ms < 4; ++ms) {
#pragma unroll
    for (int cc = 0; cc < 4; ++cc) {
      int n = wave * 64 + cc * 16 + lrow;
      float bcv = bc[n];
#pragma unroll
      for (int r = 0; r < 8; ++r) {
        int mr = mrow0 + ms * 16 + lhi * 8 + r;          // C layout: lanes16-31 -> M=r+8
        if (mr < NN) xl[(size_t)mr * HID + n] = acc[ms][cc][r] + bcv;
      }
    }
  }
}

// ================= K4: a_s, a_d per node (wave per node) =================
__global__ void k_row_dots(const float* __restrict__ xl, const float* __restrict__ att_src,
                           const float* __restrict__ att_dst,
                           float* __restrict__ a_s, float* __restrict__ a_d) {
  int wid = (blockIdx.x * blockDim.x + threadIdx.x) >> 5;
  int l = threadIdx.x & 31;
  if (wid >= NN) return;
  const float* row = xl + (size_t)wid * HID;
  float s1 = 0.f, s2 = 0.f;
#pragma unroll
  for (int j = 0; j < 8; ++j) {
    int c = l + 32 * j;
    float v = row[c];
    s1 += v * att_src[c];
    s2 += v * att_dst[c];
  }
  for (int off = 16; off; off >>= 1) { s1 += __shfl_xor(s1, off); s2 += __shfl_xor(s2, off); }
  if (l == 0) { a_s[wid] = s1; a_d[wid] = s2; }
}

// ================= K5: edge pass B (wave per edge) =================
__global__ void k_edge_b(const float* __restrict__ edge_attr, const int* __restrict__ ei,
                         const float* __restrict__ u, const float* __restrict__ c0,
                         const float* __restrict__ a_s, const float* __restrict__ a_d,
                         float* __restrict__ s_attr, float* __restrict__ deg,
                         float* __restrict__ logit_e, int* __restrict__ mbuf) {
  int e = (blockIdx.x * blockDim.x + threadIdx.x) >> 5;
  int l = threadIdx.x & 31;
  if (e >= EE) return;
  int src = ei[e], dst = ei[EE + e];
  float2 ea = *reinterpret_cast<const float2*>(edge_attr + (size_t)e * EDGE_C + 2 * l);
  float part = ea.x * u[2 * l] + ea.y * u[2 * l + 1];
  for (int off = 16; off; off >>= 1) part += __shfl_xor(part, off);
  atomicAdd(&s_attr[(size_t)dst * EDGE_C + 2 * l],     ea.x);
  atomicAdd(&s_attr[(size_t)dst * EDGE_C + 2 * l + 1], ea.y);
  if (l == 0) {
    atomicAdd(&deg[dst], 1.0f);
    float lg = leaky(a_s[src] + a_d[dst] + part + c0[0]);
    logit_e[e] = lg;
    atomicMax(&mbuf[dst], fmap_ord(lg));
  }
}

// ================= K6: node pass C (wave per node) =================
// self-loop logit, final max, den init, out_acc init
__global__ void k_node_c(const float* __restrict__ s_attr, const float* __restrict__ deg,
                         const float* __restrict__ u, const float* __restrict__ c0,
                         const float* __restrict__ a_s, const float* __restrict__ a_d,
                         const float* __restrict__ xl,
                         float* __restrict__ mbuf,           // int in, float out
                         float* __restrict__ den, float* __restrict__ out_acc) {
  int n = (blockIdx.x * blockDim.x + threadIdx.x) >> 5;
  int l = threadIdx.x & 31;
  if (n >= NN) return;
  float2 sa = *reinterpret_cast<const float2*>(s_attr + (size_t)n * EDGE_C + 2 * l);
  float part = sa.x * u[2 * l] + sa.y * u[2 * l + 1];
  for (int off = 16; off; off >>= 1) part += __shfl_xor(part, off);
  float d = deg[n];
  float ae = (d > 0.f) ? (part / d + c0[0]) : 0.f;
  float sl = leaky(a_s[n] + a_d[n] + ae);
  float mf = 0.f;
  if (l == 0) {
    float me = funmap_ord(reinterpret_cast<const int*>(mbuf)[n]);
    mf = fmaxf(me, sl);
    mbuf[n] = mf;                                    // now holds float max
  }
  mf = __shfl(mf, 0);
  float p = __expf(sl - mf);
  if (l == 0) den[n] = p;
  const float* row = xl + (size_t)n * HID;
  float* orow = out_acc + (size_t)n * HID;
#pragma unroll
  for (int j = 0; j < 8; ++j) { int c = l + 32 * j; orow[c] = p * row[c]; }
}

// ================= K7: edge pass D (wave per edge): scatter =================
__global__ void k_edge_d(const int* __restrict__ ei, const float* __restrict__ logit_e,
                         const float* __restrict__ mbuf, const float* __restrict__ xl,
                         float* __restrict__ den, float* __restrict__ out_acc) {
  int e = (blockIdx.x * blockDim.x + threadIdx.x) >> 5;
  int l = threadIdx.x & 31;
  if (e >= EE) return;
  int src = ei[e], dst = ei[EE + e];
  float p = __expf(logit_e[e] - mbuf[dst]);
  if (l == 0) atomicAdd(&den[dst], p);
  const float* row = xl + (size_t)src * HID;
  float* orow = out_acc + (size_t)dst * HID;
#pragma unroll
  for (int j = 0; j < 8; ++j) {
    int c = l + 32 * j;
    atomicAdd(&orow[c], p * row[c]);
  }
}

// ================= K8: normalize by den + b_gat, BN partial stats =================
__global__ void k_finish_bn(float* __restrict__ out_acc, const float* __restrict__ den,
                            const float* __restrict__ b_gat,
                            float* __restrict__ bn_sum, float* __restrict__ bn_sumsq) {
  int t = threadIdx.x;                                // channel
  size_t base = (size_t)blockIdx.x * 32;
  float s = 0.f, s2 = 0.f;
  float bg = b_gat[t];
  for (int i = 0; i < 32; ++i) {
    size_t n = base + i;
    if (n < NN) {
      float v = out_acc[n * HID + t] / den[n] + bg;
      out_acc[n * HID + t] = v;
      s += v; s2 += v * v;
    }
  }
  atomicAdd(&bn_sum[t], s);
  atomicAdd(&bn_sumsq[t], s2);
}

// ================= K9: BN finalize =================
__global__ void k_bn_final(const float* bn_sum, const float* bn_sumsq,
                           float* bn_mu, float* bn_rstd) {
  int t = threadIdx.x;
  float mu = bn_sum[t] / (float)NN;
  float var = bn_sumsq[t] / (float)NN - mu * mu;
  bn_mu[t] = mu;
  bn_rstd[t] = rsqrtf(var + EPS);
}

// ================= K10: xt = relu(BN(out[:SEQ])) =================
__global__ void k_make_xt(const float* __restrict__ out_acc, const float* __restrict__ mu,
                          const float* __restrict__ rstd, const float* __restrict__ g,
                          const float* __restrict__ b, float* __restrict__ xt) {
  int r = blockIdx.x, t = threadIdx.x;
  float v = (out_acc[(size_t)r * HID + t] - mu[t]) * rstd[t] * g[t] + b[t];
  xt[r * HID + t] = fmaxf(v, 0.f);
}

// ================= K11: qkv =================
__global__ void k_qkv(const float* __restrict__ xt, const float* __restrict__ W,
                      const float* __restrict__ b, float* __restrict__ qkv) {
  int r = blockIdx.x, t = threadIdx.x;
  __shared__ float sx[HID];
  sx[t] = xt[r * HID + t];
  __syncthreads();
#pragma unroll
  for (int k = 0; k < 3; ++k) {
    int o = t + k * HID;
    float a = b[o];
    for (int j = 0; j < HID; ++j) a += sx[j] * W[j * (3 * HID) + o];
    qkv[r * (3 * HID) + o] = a;
  }
}

// ================= K12: attention per (q, head) =================
__global__ void k_attn(const float* __restrict__ qkv, float* __restrict__ ctx) {
  int q = blockIdx.x >> 1, h = blockIdx.x & 1;
  int t = threadIdx.x;                                // 128 threads
  __shared__ float sc[SEQ];
  __shared__ float sinv;
  const float scale = 0.088388347648318447f;          // 1/sqrt(128)
  if (t < SEQ) {
    float a = 0.f;
    const float* qv = qkv + q * (3 * HID) + h * HEAD_D;
    const float* kv = qkv + t * (3 * HID) + HID + h * HEAD_D;
    for (int d = 0; d < HEAD_D; ++d) a += qv[d] * kv[d];
    sc[t] = a * scale;
  }
  __syncthreads();
  if (t == 0) {
    float mx = -1e30f;
    for (int k = 0; k < SEQ; ++k) mx = fmaxf(mx, sc[k]);
    float ssum = 0.f;
    for (int k = 0; k < SEQ; ++k) { sc[k] = __expf(sc[k] - mx); ssum += sc[k]; }
    sinv = 1.f / ssum;
  }
  __syncthreads();
  float a = 0.f;
  for (int k = 0; k < SEQ; ++k) a += sc[k] * qkv[k * (3 * HID) + 2 * HID + h * HEAD_D + t];
  ctx[q * HID + h * HEAD_D + t] = a * sinv;
}

// ================= K13: attn_out + residual + LN1 =================
__global__ void k_attnout_ln1(const float* __restrict__ ctx, const float* __restrict__ W,
                              const float* __restrict__ b, const float* __restrict__ xt,
                              const float* __restrict__ g, const float* __restrict__ bb,
                              float* __restrict__ y1) {
  int r = blockIdx.x, t = threadIdx.x;
  __shared__ float sx[HID];
  __shared__ float sred[HID];
  sx[t] = ctx[r * HID + t];
  __syncthreads();
  float a = b[t];
  for (int j = 0; j < HID; ++j) a += sx[j] * W[j * HID + t];
  float res = xt[r * HID + t] + a;
  sred[t] = res; __syncthreads();
  for (int s = 128; s > 0; s >>= 1) { if (t < s) sred[t] += sred[t + s]; __syncthreads(); }
  float mu = sred[0] / (float)HID; __syncthreads();
  float d = res - mu;
  sred[t] = d * d; __syncthreads();
  for (int s = 128; s > 0; s >>= 1) { if (t < s) sred[t] += sred[t + s]; __syncthreads(); }
  float var = sred[0] / (float)HID;
  y1[r * HID + t] = d * rsqrtf(var + EPS) * g[t] + bb[t];
}

// ================= K14: FFN layer 1 =================
__global__ void k_ffn1(const float* __restrict__ y1, const float* __restrict__ W,
                       const float* __restrict__ b, float* __restrict__ ff1) {
  int r = blockIdx.x, t = threadIdx.x;
  __shared__ float sy[HID];
  sy[t] = y1[r * HID + t];
  __syncthreads();
#pragma unroll
  for (int k = 0; k < 2; ++k) {
    int o = t + k * HID;
    float a = b[o];
    for (int j = 0; j < HID; ++j) a += sy[j] * W[j * (2 * HID) + o];
    ff1[r * (2 * HID) + o] = fmaxf(a, 0.f);
  }
}

// ================= K15: FFN2 + residual + LN2 + LNf + pool =================
__global__ void k_ffn2_lnf(const float* __restrict__ ff1, const float* __restrict__ W2,
                           const float* __restrict__ b2, const float* __restrict__ y1,
                           const float* __restrict__ g2, const float* __restrict__ bb2,
                           const float* __restrict__ gf, const float* __restrict__ bf,
                           float* __restrict__ pooled) {
  int r = blockIdx.x, t = threadIdx.x;
  __shared__ float sff[2 * HID];
  __shared__ float sred[HID];
  sff[t] = ff1[r * (2 * HID) + t];
  sff[t + HID] = ff1[r * (2 * HID) + t + HID];
  __syncthreads();
  float a = b2[t];
  for (int j = 0; j < 2 * HID; ++j) a += sff[j] * W2[j * HID + t];
  float res = y1[r * HID + t] + a;
  // LN2
  sred[t] = res; __syncthreads();
  for (int s = 128; s > 0; s >>= 1) { if (t < s) sred[t] += sred[t + s]; __syncthreads(); }
  float mu = sred[0] / (float)HID; __syncthreads();
  float d = res - mu;
  sred[t] = d * d; __syncthreads();
  for (int s = 128; s > 0; s >>= 1) { if (t < s) sred[t] += sred[t + s]; __syncthreads(); }
  float var = sred[0] / (float)HID; __syncthreads();
  float z = d * rsqrtf(var + EPS) * g2[t] + bb2[t];
  // LNf
  sred[t] = z; __syncthreads();
  for (int s = 128; s > 0; s >>= 1) { if (t < s) sred[t] += sred[t + s]; __syncthreads(); }
  float mu2 = sred[0] / (float)HID; __syncthreads();
  float d2 = z - mu2;
  sred[t] = d2 * d2; __syncthreads();
  for (int s = 128; s > 0; s >>= 1) { if (t < s) sred[t] += sred[t + s]; __syncthreads(); }
  float var2 = sred[0] / (float)HID;
  float outv = d2 * rsqrtf(var2 + EPS) * gf[t] + bf[t];
  atomicAdd(&pooled[t], outv * (1.0f / (float)SEQ));
}

// ================= K16: final projection =================
__global__ void k_final(const float* __restrict__ pooled, const float* __restrict__ W,
                        const float* __restrict__ b, float* __restrict__ out) {
  int o = threadIdx.x;                                // 128 threads
  float a = b[o];
  for (int c = 0; c < HID; ++c) a += pooled[c] * W[c * OUT_C + o];
  out[o] = a;
}

// =====================================================================
extern "C" void kernel_launch(void* const* d_in, const int* in_sizes, int n_in,
                              void* d_out, int out_size, void* d_ws, size_t ws_size,
                              hipStream_t stream) {
  const float* x          = (const float*)d_in[0];
  const float* edge_attr  = (const float*)d_in[1];
  const float* w_node     = (const float*)d_in[2];
  const float* b_node     = (const float*)d_in[3];
  const float* w_edge     = (const float*)d_in[4];
  const float* b_edge     = (const float*)d_in[5];
  const float* w_gat      = (const float*)d_in[6];
  const float* w_gat_edge = (const float*)d_in[7];
  const float* att_src    = (const float*)d_in[8];
  const float* att_dst    = (const float*)d_in[9];
  const float* att_edge   = (const float*)d_in[10];
  const float* b_gat      = (const float*)d_in[11];
  const float* bn_gamma   = (const float*)d_in[12];
  const float* bn_beta    = (const float*)d_in[13];
  const float* attn_in_w  = (const float*)d_in[14];
  const float* attn_in_b  = (const float*)d_in[15];
  const float* attn_out_w = (const float*)d_in[16];
  const float* attn_out_b = (const float*)d_in[17];
  const float* ln1_g      = (const float*)d_in[18];
  const float* ln1_b      = (const float*)d_in[19];
  const float* ffn_w1     = (const float*)d_in[20];
  const float* ffn_b1     = (const float*)d_in[21];
  const float* ffn_w2     = (const float*)d_in[22];
  const float* ffn_b2     = (const float*)d_in[23];
  const float* ln2_g      = (const float*)d_in[24];
  const float* ln2_b      = (const float*)d_in[25];
  const float* lnf_g      = (const float*)d_in[26];
  const float* lnf_b      = (const float*)d_in[27];
  const float* w_out      = (const float*)d_in[28];
  const float* b_out      = (const float*)d_in[29];
  const int*   ei         = (const int*)d_in[30];

  // ---- workspace bump allocator ----
  char* wp = (char*)d_ws;
  auto alloc = [&](size_t bytes) -> void* {
    void* r = (void*)wp;
    wp += (bytes + 255) & ~(size_t)255;
    return r;
  };
  float*          xl       = (float*)alloc((size_t)NN * HID * 4);
  float*          out_acc  = (float*)alloc((size_t)NN * HID * 4);
  unsigned short* x_bf16   = (unsigned short*)alloc((size_t)NN * IN_C * 2);
  float*          s_attr   = (float*)alloc((size_t)NN * EDGE_C * 4);
  float*          logit_e  = (float*)alloc((size_t)EE * 4);
  float*          deg      = (float*)alloc((size_t)NN * 4);
  float*          a_s      = (float*)alloc((size_t)NN * 4);
  float*          a_d      = (float*)alloc((size_t)NN * 4);
  float*          mbuf     = (float*)alloc((size_t)NN * 4);   // int map, then float max
  float*          den      = (float*)alloc((size_t)NN * 4);
  unsigned short* bpack    = (unsigned short*)alloc((size_t)IN_C * HID * 2);
  float*          bc       = (float*)alloc(HID * 4);
  float*          vbuf     = (float*)alloc(HID * 4);
  float*          ubuf     = (float*)alloc(EDGE_C * 4);
  float*          c0buf    = (float*)alloc(4);
  float*          bn_sum   = (float*)alloc(HID * 4);
  float*          bn_sumsq = (float*)alloc(HID * 4);
  float*          bn_mu    = (float*)alloc(HID * 4);
  float*          bn_rstd  = (float*)alloc(HID * 4);
  float*          xt       = (float*)alloc((size_t)SEQ * HID * 4);
  float*          qkv      = (float*)alloc((size_t)SEQ * 3 * HID * 4);
  float*          ctx      = (float*)alloc((size_t)SEQ * HID * 4);
  float*          y1       = (float*)alloc((size_t)SEQ * HID * 4);
  float*          ff1      = (float*)alloc((size_t)SEQ * 2 * HID * 4);
  float*          pooled   = (float*)alloc(HID * 4);

  // ---- pipeline ----
  k_precompute<<<1, 256, 0, stream>>>(w_node, b_node, w_edge, b_edge, w_gat, w_gat_edge,
                                      att_edge, vbuf, ubuf, c0buf, bc, bpack);
  k_cvt_x<<<(NN * IN_C + 255) / 256, 256, 0, stream>>>(x, x_bf16, NN * IN_C);
  k_init<<<(NN * EDGE_C + 255) / 256, 256, 0, stream>>>(s_attr, deg, (int*)mbuf,
                                                        bn_sum, bn_sumsq, pooled);
  k_gemm_xl<<<(NN + 63) / 64, 128, 0, stream>>>(x_bf16, bpack, bc, xl);
  k_row_dots<<<(NN * 32 + 255) / 256, 256, 0, stream>>>(xl, att_src, att_dst, a_s, a_d);
  k_edge_b<<<(EE * 32 + 255) / 256, 256, 0, stream>>>(edge_attr, ei, ubuf, c0buf,
                                                      a_s, a_d, s_attr, deg, logit_e, (int*)mbuf);
  k_node_c<<<(NN * 32 + 255) / 256, 256, 0, stream>>>(s_attr, deg, ubuf, c0buf,
                                                      a_s, a_d, xl, mbuf, den, out_acc);
  k_edge_d<<<(EE * 32 + 255) / 256, 256, 0, stream>>>(ei, logit_e, mbuf, xl, den, out_acc);
  k_finish_bn<<<(NN + 31) / 32, 256, 0, stream>>>(out_acc, den, b_gat, bn_sum, bn_sumsq);
  k_bn_final<<<1, 256, 0, stream>>>(bn_sum, bn_sumsq, bn_mu, bn_rstd);
  k_make_xt<<<SEQ, 256, 0, stream>>>(out_acc, bn_mu, bn_rstd, bn_gamma, bn_beta, xt);
  k_qkv<<<SEQ, 256, 0, stream>>>(xt, attn_in_w, attn_in_b, qkv);
  k_attn<<<SEQ * HEADS, 128, 0, stream>>>(qkv, ctx);
  k_attnout_ln1<<<SEQ, 256, 0, stream>>>(ctx, attn_out_w, attn_out_b, xt, ln1_g, ln1_b, y1);
  k_ffn1<<<SEQ, 256, 0, stream>>>(y1, ffn_w1, ffn_b1, ff1);
  k_ffn2_lnf<<<SEQ, 256, 0, stream>>>(ff1, ffn_w2, ffn_b2, y1, ln2_g, ln2_b, lnf_g, lnf_b, pooled);
  k_final<<<1, 128, 0, stream>>>(pooled, w_out, b_out, (float*)d_out);
}
